// TransitionDownBlock_26121991094973
// MI455X (gfx1250) — matrix operand order repaired and checked by reading, not compiled, not run on hardware
//
#include <hip/hip_runtime.h>
#include <hip/hip_bf16.h>
#include <math.h>

// ---------------------------------------------------------------------------
// TransitionDown block (FPS + kNN + mlp1/BN + local transformer + mlp2/BN +
// maxpool) for MI455X (gfx1250, wave32).  All dense contractions run on
// v_wmma_f32_16x16x32_f16; weights are pre-swizzled into fragment-native
// tile order so every B-fragment load is one contiguous 32B load per lane.
// ---------------------------------------------------------------------------

typedef __attribute__((ext_vector_type(16))) _Float16 v16h;
typedef __attribute__((ext_vector_type(8)))  float    v8f;
typedef __attribute__((ext_vector_type(4)))  unsigned int v4u;
typedef __attribute__((ext_vector_type(4)))  int v4i;
typedef __attribute__((ext_vector_type(8)))  int v8i;

#if defined(__AMDGCN__) && __has_builtin(__builtin_amdgcn_tensor_load_to_lds)
#define HAVE_TDM 1
#else
#define HAVE_TDM 0
#endif

#define BATCH   8
#define NPTS    4096
#define NPOINT  1024
#define KNN     16
#define CIN     64
#define DMODEL  512
#define COUT    128
#define NGROUP  (BATCH * NPOINT)          // 8192
#define NROWS   (NGROUP * KNN)            // 131072 rows of the flattened GEMMs

__device__ __forceinline__ v8f wmma_f16(v16h a, v16h b, v8f c) {
  return __builtin_amdgcn_wmma_f32_16x16x32_f16(false, a, false, b, (short)0, c,
                                                false, false);
}

// ---- CDNA5 WMMA fragment helpers (wave32) ----------------------------------
// A: 16x32 f16 row-major (ld elems): lanes 0-15 row=lane, K=[0..7]+[16..23];
//    lanes 16-31 K=[8..15]+[24..31].  Vectorizes to 2x ds_load_b128 from LDS.
__device__ __forceinline__ v16h load_a(const _Float16* src, int ld) {
  int lane = threadIdx.x & 31;
  int row  = lane & 15;
  int klo  = (lane < 16) ? 0 : 8;
  int khi  = (lane < 16) ? 16 : 24;
  v16h a;
#pragma unroll
  for (int i = 0; i < 8; ++i) a[i] = src[row * ld + klo + i];
#pragma unroll
  for (int i = 0; i < 8; ++i) a[8 + i] = src[row * ld + khi + i];
  return a;
}
// A with K-depth 16 (upper 16 K zero-padded)
__device__ __forceinline__ v16h load_a_k16(const _Float16* src, int ld) {
  int lane = threadIdx.x & 31;
  int row  = lane & 15;
  int klo  = (lane < 16) ? 0 : 8;
  v16h a = {};
#pragma unroll
  for (int i = 0; i < 8; ++i) a[i] = src[row * ld + klo + i];
  return a;
}
// B from a pre-swizzled tile: lane's 16 halves are contiguous (32B coalesced).
__device__ __forceinline__ v16h load_b_swz(const _Float16* tile) {
  int lane = threadIdx.x & 31;
  return *(const v16h*)(tile + lane * 16);
}
// B 16x16 [K][N] in LDS, K>=16 zero
__device__ __forceinline__ v16h load_b_k16(const _Float16* src, int ld) {
  int lane = threadIdx.x & 31;
  int n    = lane & 15;
  v16h b = {};
  int kb   = (lane < 16) ? 0 : 8;
#pragma unroll
  for (int i = 0; i < 8; ++i) b[i] = src[(kb + i) * ld + n];
  return b;
}
// B = transpose of a 16x16 LDS tile (element (k,n) = src[n*ld+k]), K>=16 zero
__device__ __forceinline__ v16h load_bT_k16(const _Float16* src, int ld) {
  int lane = threadIdx.x & 31;
  int n    = lane & 15;
  v16h b = {};
  int kb   = (lane < 16) ? 0 : 8;
#pragma unroll
  for (int i = 0; i < 8; ++i) b[i] = src[n * ld + kb + i];
  return b;
}
// C/D: lanes 0-15: N=lane, M=r; lanes 16-31: N=lane-16, M=8+r
__device__ __forceinline__ void store_c_h(v8f c, _Float16* dst, int ld) {
  int lane = threadIdx.x & 31;
  int n    = lane & 15;
  int mb   = (lane < 16) ? 0 : 8;
#pragma unroll
  for (int r = 0; r < 8; ++r) dst[(mb + r) * ld + n] = (_Float16)c[r];
}
__device__ __forceinline__ void store_c_f(v8f c, float* dst, int ld) {
  int lane = threadIdx.x & 31;
  int n    = lane & 15;
  int mb   = (lane < 16) ? 0 : 8;
#pragma unroll
  for (int r = 0; r < 8; ++r) dst[(mb + r) * ld + n] = c[r];
}

// ---------------------------------------------------------------------------
// utility kernels
// ---------------------------------------------------------------------------
__global__ void zero_f32(float* p, int n) {
  int i = blockIdx.x * blockDim.x + threadIdx.x;
  if (i < n) p[i] = 0.f;
}

// W [N][K] f32  ->  fragment-swizzled f16 tiles.
// Tile t = ks*(N/16)+nt holds the 32x16 B tile (K rows ks*32.., cols nt*16..),
// laid out as 32 lanes x 16 contiguous halves in exact v_wmma B order.
__global__ void swizzle_weights(const float* __restrict__ w,
                                _Float16* __restrict__ out, int N, int K) {
  int i = blockIdx.x * blockDim.x + threadIdx.x;
  if (i >= N * K) return;
  int half = i & 15;
  int lane = (i >> 4) & 31;
  int tile = i >> 9;
  int NT = N >> 4;
  int ks = tile / NT, nt = tile - ks * NT;
  int n = nt * 16 + (lane & 15);
  int k = ks * 32 + ((lane < 16) ? 0 : 8) + (half & 7) + ((half >> 3) << 4);
  out[i] = (_Float16)w[(size_t)n * K + k];
}

// ---------------------------------------------------------------------------
// 1) deterministic farthest point sampling: 1 block per batch
// ---------------------------------------------------------------------------
__global__ __launch_bounds__(256) void fps_kernel(const float* __restrict__ xyz,
                                                  int* __restrict__ fps_idx,
                                                  float* __restrict__ out_xyz) {
  int b = blockIdx.x;
  const float* X = xyz + (size_t)b * NPTS * 3;
  __shared__ float sdist[NPTS];
  __shared__ float rv[256];
  __shared__ int   ri[256];
  __shared__ int   cur;
  __shared__ float cx, cy, cz;
  for (int i = threadIdx.x; i < NPTS; i += 256) sdist[i] = 1e10f;
  if (threadIdx.x == 0) cur = 0;
  __syncthreads();
  for (int s = 0; s < NPOINT; ++s) {
    if (threadIdx.x == 0) {
      int f = cur;
      fps_idx[b * NPOINT + s] = f;
      float a0 = X[f * 3 + 0], a1 = X[f * 3 + 1], a2 = X[f * 3 + 2];
      cx = a0; cy = a1; cz = a2;
      out_xyz[(size_t)(b * NPOINT + s) * 3 + 0] = a0;
      out_xyz[(size_t)(b * NPOINT + s) * 3 + 1] = a1;
      out_xyz[(size_t)(b * NPOINT + s) * 3 + 2] = a2;
    }
    __syncthreads();
    float best = -1.f;
    int   bi   = NPTS;
    for (int i = threadIdx.x; i < NPTS; i += 256) {
      float dx = X[i * 3 + 0] - cx, dy = X[i * 3 + 1] - cy, dz = X[i * 3 + 2] - cz;
      float d  = dx * dx + dy * dy + dz * dz;
      float nd = fminf(sdist[i], d);
      sdist[i] = nd;
      if (nd > best || (nd == best && i < bi)) { best = nd; bi = i; }
    }
    rv[threadIdx.x] = best; ri[threadIdx.x] = bi;
    __syncthreads();
    for (int off = 128; off > 0; off >>= 1) {
      if (threadIdx.x < off) {
        float ov = rv[threadIdx.x + off]; int oi = ri[threadIdx.x + off];
        if (ov > rv[threadIdx.x] || (ov == rv[threadIdx.x] && oi < ri[threadIdx.x])) {
          rv[threadIdx.x] = ov; ri[threadIdx.x] = oi;
        }
      }
      __syncthreads();
    }
    if (threadIdx.x == 0) cur = ri[0];
    __syncthreads();
  }
}

// ---------------------------------------------------------------------------
// 2) kNN (16 smallest sq-distances, ascending, tie->lowest index) + gather f16
// ---------------------------------------------------------------------------
__global__ __launch_bounds__(256) void knn_gather_kernel(
    const float* __restrict__ xyz, const float* __restrict__ feat,
    const int* __restrict__ fps_idx, _Float16* __restrict__ Afeat) {
  int g = blockIdx.x;
  int b = g >> 10;
  const float* X = xyz + (size_t)b * NPTS * 3;
  __shared__ float sd[NPTS];
  __shared__ float rv[256];
  __shared__ int   ri[256];
  __shared__ int   sel[KNN];
  int   c  = fps_idx[g];
  float cx = X[c * 3 + 0], cy = X[c * 3 + 1], cz = X[c * 3 + 2];
  for (int i = threadIdx.x; i < NPTS; i += 256) {
    float dx = X[i * 3 + 0] - cx, dy = X[i * 3 + 1] - cy, dz = X[i * 3 + 2] - cz;
    sd[i] = dx * dx + dy * dy + dz * dz;
  }
  __syncthreads();
  for (int kk = 0; kk < KNN; ++kk) {
    float best = 1e30f; int bi = NPTS;
    for (int i = threadIdx.x; i < NPTS; i += 256) {
      float v = sd[i];
      if (v < best || (v == best && i < bi)) { best = v; bi = i; }
    }
    rv[threadIdx.x] = best; ri[threadIdx.x] = bi;
    __syncthreads();
    for (int off = 128; off > 0; off >>= 1) {
      if (threadIdx.x < off) {
        float ov = rv[threadIdx.x + off]; int oi = ri[threadIdx.x + off];
        if (ov < rv[threadIdx.x] || (ov == rv[threadIdx.x] && oi < ri[threadIdx.x])) {
          rv[threadIdx.x] = ov; ri[threadIdx.x] = oi;
        }
      }
      __syncthreads();
    }
    if (threadIdx.x == 0) { sel[kk] = ri[0]; sd[ri[0]] = 1e30f; }
    __syncthreads();
  }
  for (int t = threadIdx.x; t < KNN * CIN; t += 256) {
    int kk = t >> 6, ch = t & 63;
    Afeat[(size_t)g * KNN * CIN + t] =
        (_Float16)feat[((size_t)b * NPTS + sel[kk]) * CIN + ch];
  }
}

// ---------------------------------------------------------------------------
// 3) mlp1: X1 = grouped @ w1^T + b1 (per-group 16x64x64 WMMA) + BN1 stats
// ---------------------------------------------------------------------------
__global__ __launch_bounds__(128) void mlp1_kernel(
    const _Float16* __restrict__ Afeat, const _Float16* __restrict__ w1h,
    const float* __restrict__ b1, _Float16* __restrict__ X1h,
    float* __restrict__ stats1) {
  int g = blockIdx.x, tid = threadIdx.x;
  __shared__ _Float16 sA[16 * 64];
  __shared__ float ssum[64], ssq[64];
  for (int t = tid; t < 16 * 64; t += 128) sA[t] = Afeat[(size_t)g * 1024 + t];
  if (tid < 64) { ssum[tid] = 0.f; ssq[tid] = 0.f; }
  __syncthreads();
  int wave = tid >> 5, lane = tid & 31;
  v8f acc = {};
#pragma unroll
  for (int ks = 0; ks < 2; ++ks)
    acc = wmma_f16(load_a(sA + ks * 32, 64),
                   load_b_swz(w1h + ((size_t)(ks * 4 + wave) << 9)), acc);
  int n = wave * 16 + (lane & 15), mb = (lane < 16) ? 0 : 8;
  float bias = b1[n];
  float lsum = 0.f, lsq = 0.f;
#pragma unroll
  for (int r = 0; r < 8; ++r) {
    float v = acc[r] + bias;
    X1h[(size_t)g * 1024 + (mb + r) * 64 + n] = (_Float16)v;
    lsum += v; lsq += v * v;
  }
  atomicAdd(&ssum[n], lsum);
  atomicAdd(&ssq[n], lsq);
  __syncthreads();
  if (tid < 64) {
    atomicAdd(&stats1[tid], ssum[tid]);
    atomicAdd(&stats1[64 + tid], ssq[tid]);
  }
}

__global__ void bn_finalize(const float* __restrict__ stats,
                            const float* __restrict__ gamma,
                            const float* __restrict__ beta,
                            float* __restrict__ scale, float* __restrict__ bias,
                            int C, float inv_count) {
  int c = threadIdx.x;
  if (c < C) {
    float mu  = stats[c] * inv_count;
    float var = stats[C + c] * inv_count - mu * mu;
    float sc  = gamma[c] * rsqrtf(var + 1e-5f);
    scale[c] = sc;
    bias[c]  = beta[c] - mu * sc;
  }
}

__global__ void bn1_apply(const _Float16* __restrict__ X1h,
                          const float* __restrict__ scale,
                          const float* __restrict__ bias,
                          _Float16* __restrict__ Xbn) {
  size_t i = (size_t)blockIdx.x * blockDim.x + threadIdx.x;
  if (i < (size_t)NROWS * CIN) {
    int c = (int)(i & 63);
    float v = (float)X1h[i] * scale[c] + bias[c];
    Xbn[i] = (_Float16)fmaxf(v, 0.f);
  }
}

// ---------------------------------------------------------------------------
// 4) fused transformer block per group (256 thr = 8 waves)
// ---------------------------------------------------------------------------
__global__ __launch_bounds__(256) void attn_kernel(
    const _Float16* __restrict__ Xbn, const _Float16* __restrict__ fc1h,
    const float* __restrict__ fc1_b, const _Float16* __restrict__ wqh,
    const _Float16* __restrict__ wkh, const _Float16* __restrict__ wvh,
    const _Float16* __restrict__ fc2h, const float* __restrict__ fc2_b,
    const _Float16* __restrict__ w2h, const float* __restrict__ b2,
    _Float16* __restrict__ Yh, float* __restrict__ stats2) {
  int g = blockIdx.x, tid = threadIdx.x;
  int wave = tid >> 5, lane = tid & 31;
  __shared__ _Float16 xg[16 * 64];        // bn-relu'd mlp1 output (A + residual)
  __shared__ _Float16 hbuf[16 * DMODEL];  // fc1 output
  __shared__ _Float16 resbuf[16 * DMODEL];
  __shared__ _Float16 stage[8][16 * 16];  // per-wave q_c / v_c staging
  __shared__ _Float16 kst[8][16 * 16];    // per-wave k_c staging
  __shared__ float    sc_part[8][16 * 16];
  __shared__ float    scores[16 * 16];
  __shared__ _Float16 attnw[16 * 32];     // K zero-padded to 32
  __shared__ _Float16 x2[16 * 64];
  __shared__ float    ssum2[128], ssq2[128];

#if HAVE_TDM
  // Tensor Data Mover: DMA the 16x64 f16 group tile straight into LDS.
  if (wave == 0) {
    unsigned lds_addr = (unsigned)(uintptr_t)(void*)xg;
    unsigned long long ga =
        (unsigned long long)(uintptr_t)(Xbn + (size_t)g * 1024);
    v4u g0;
    g0.x = 1u;                                     // count=1 (valid descriptor)
    g0.y = lds_addr;                               // D#.lds_addr
    g0.z = (unsigned)(ga & 0xffffffffu);           // D#.global_addr[31:0]
    g0.w = (unsigned)((ga >> 32) & 0x01ffffffu) | (2u << 30);  // addr hi, type=2
    v8i g1;
    g1[0] = (int)(1u << 16);                       // data_size=2 bytes
    g1[1] = (int)((1024u & 0xffffu) << 16);        // tensor_dim0 = 1024 (lo16)
    g1[2] = (int)((1024u >> 16) | (1u << 16));     // dim0 hi | tensor_dim1=1 lo
    g1[3] = (int)(1024u << 16);                    // dim1 hi=0 | tile_dim0=1024
    g1[4] = 1;                                     // tile_dim1=1, tile_dim2=0
    g1[5] = 1024;                                  // tensor_dim0_stride lo32
    g1[6] = 0;
    g1[7] = 0;
    v4i z4 = {0, 0, 0, 0};
#if __clang_major__ >= 23
    v8i z8 = {0, 0, 0, 0, 0, 0, 0, 0};
    __builtin_amdgcn_tensor_load_to_lds(g0, g1, z4, z4, z8, 0);
#else
    __builtin_amdgcn_tensor_load_to_lds(g0, g1, z4, z4, 0);
#endif
    __builtin_amdgcn_s_wait_tensorcnt(0);
  }
#else
  for (int t = tid; t < 16 * 64; t += 256) xg[t] = Xbn[(size_t)g * 1024 + t];
#endif
  for (int t = tid; t < 16 * 32; t += 256) attnw[t] = (_Float16)0.f;
  if (tid < 128) { ssum2[tid] = 0.f; ssq2[tid] = 0.f; }
  __syncthreads();

  // ---- phase 1: h = xg @ fc1^T + fc1_b  (M=16, N=512, K=64) ----
  for (int nt = wave; nt < 32; nt += 8) {
    v8f acc = {};
#pragma unroll
    for (int ks = 0; ks < 2; ++ks)
      acc = wmma_f16(load_a(xg + ks * 32, 64),
                     load_b_swz(fc1h + ((size_t)(ks * 32 + nt) << 9)), acc);
    int n = nt * 16 + (lane & 15), mb = (lane < 16) ? 0 : 8;
    float bias = fc1_b[n];
#pragma unroll
    for (int r = 0; r < 8; ++r)
      hbuf[(mb + r) * DMODEL + n] = (_Float16)(acc[r] + bias);
  }
  __syncthreads();

  // ---- phase 2: scores = (h Wq^T)(h Wk^T)^T accumulated per 16-dim chunk ----
  v8f sacc = {};
  for (int j = 0; j < 4; ++j) {
    int chunk = wave * 4 + j;
    __builtin_prefetch(wqh + ((size_t)chunk << 9), 0, 1);
    __builtin_prefetch(wkh + ((size_t)chunk << 9), 0, 1);
    v8f qa = {}, ka = {};
    for (int ks = 0; ks < 16; ++ks) {
      v16h afr = load_a(hbuf + ks * 32, DMODEL);
      qa = wmma_f16(afr, load_b_swz(wqh + ((size_t)(ks * 32 + chunk) << 9)), qa);
      ka = wmma_f16(afr, load_b_swz(wkh + ((size_t)(ks * 32 + chunk) << 9)), ka);
    }
    store_c_h(qa, stage[wave], 16);
    store_c_h(ka, kst[wave], 16);
    sacc = wmma_f16(load_a_k16(stage[wave], 16), load_bT_k16(kst[wave], 16), sacc);
  }
  store_c_f(sacc, sc_part[wave], 16);
  __syncthreads();
  {
    float s = 0.f;
#pragma unroll
    for (int w = 0; w < 8; ++w) s += sc_part[w][tid];
    scores[tid] = s;
  }
  __syncthreads();

  // ---- phase 3: row softmax (scale 1/sqrt(512)) ----
  if (tid < 16) {
    const float inv = 0.04419417382415922f;
    float mx = -1e30f;
    for (int n = 0; n < 16; ++n) mx = fmaxf(mx, scores[tid * 16 + n] * inv);
    float e[16], ssum = 0.f;
    for (int n = 0; n < 16; ++n) {
      e[n] = __expf(scores[tid * 16 + n] * inv - mx);
      ssum += e[n];
    }
    float rs = 1.f / ssum;
    for (int n = 0; n < 16; ++n) attnw[tid * 32 + n] = (_Float16)(e[n] * rs);
  }
  __syncthreads();

  // ---- phase 4: res = attn @ (h Wv^T), per 16-dim chunk ----
  for (int j = 0; j < 4; ++j) {
    int chunk = wave * 4 + j;
    int c0 = chunk * 16;
    v8f va = {};
    for (int ks = 0; ks < 16; ++ks)
      va = wmma_f16(load_a(hbuf + ks * 32, DMODEL),
                    load_b_swz(wvh + ((size_t)(ks * 32 + chunk) << 9)), va);
    store_c_h(va, stage[wave], 16);
    v8f ra = wmma_f16(load_a(attnw, 32), load_b_k16(stage[wave], 16), (v8f){});
    store_c_h(ra, resbuf + c0, DMODEL);
  }
  __syncthreads();

  // ---- phase 5: x2 = res @ fc2^T + fc2_b + xg  (N=64, K=512) ----
  if (wave < 4) {
    v8f acc = {};
    for (int ks = 0; ks < 16; ++ks)
      acc = wmma_f16(load_a(resbuf + ks * 32, DMODEL),
                     load_b_swz(fc2h + ((size_t)(ks * 4 + wave) << 9)), acc);
    int n = wave * 16 + (lane & 15), mb = (lane < 16) ? 0 : 8;
    float bias = fc2_b[n];
#pragma unroll
    for (int r = 0; r < 8; ++r) {
      float v = acc[r] + bias + (float)xg[(mb + r) * 64 + n];
      x2[(mb + r) * 64 + n] = (_Float16)v;
    }
  }
  __syncthreads();

  // ---- phase 6: y = x2 @ w2^T + b2 (N=128, K=64) + BN2 stats ----
  {
    v8f acc = {};
#pragma unroll
    for (int ks = 0; ks < 2; ++ks)
      acc = wmma_f16(load_a(x2 + ks * 32, 64),
                     load_b_swz(w2h + ((size_t)(ks * 8 + wave) << 9)), acc);
    int n = wave * 16 + (lane & 15), mb = (lane < 16) ? 0 : 8;
    float bias = b2[n];
    float lsum = 0.f, lsq = 0.f;
#pragma unroll
    for (int r = 0; r < 8; ++r) {
      float v = acc[r] + bias;
      Yh[(size_t)g * (KNN * COUT) + (mb + r) * COUT + n] = (_Float16)v;
      lsum += v; lsq += v * v;
    }
    atomicAdd(&ssum2[n], lsum);
    atomicAdd(&ssq2[n], lsq);
  }
  __syncthreads();
  if (tid < 128) {
    atomicAdd(&stats2[tid], ssum2[tid]);
    atomicAdd(&stats2[128 + tid], ssq2[tid]);
  }
}

// ---------------------------------------------------------------------------
// 5) BN2 + ReLU + maxpool over neighbors
// ---------------------------------------------------------------------------
__global__ __launch_bounds__(128) void finalize_kernel(
    const _Float16* __restrict__ Yh, const float* __restrict__ scale2,
    const float* __restrict__ bias2, float* __restrict__ out_feat) {
  int g = blockIdx.x, c = threadIdx.x;
  float sc = scale2[c], bi = bias2[c];
  float best = -1e30f;
  for (int k = 0; k < KNN; ++k)
    best = fmaxf(best, (float)Yh[(size_t)g * (KNN * COUT) + k * COUT + c] * sc + bi);
  out_feat[(size_t)g * COUT + c] = fmaxf(best, 0.f);
}

// ---------------------------------------------------------------------------
extern "C" void kernel_launch(void* const* d_in, const int* in_sizes, int n_in,
                              void* d_out, int out_size, void* d_ws,
                              size_t ws_size, hipStream_t stream) {
  const float* xyz   = (const float*)d_in[0];
  const float* feat  = (const float*)d_in[1];
  const float* w1    = (const float*)d_in[2];
  const float* b1    = (const float*)d_in[3];
  const float* g1    = (const float*)d_in[4];
  const float* bt1   = (const float*)d_in[5];
  const float* fc1_w = (const float*)d_in[6];
  const float* fc1_b = (const float*)d_in[7];
  const float* wq    = (const float*)d_in[8];
  const float* wk    = (const float*)d_in[9];
  const float* wv    = (const float*)d_in[10];
  const float* fc2_w = (const float*)d_in[11];
  const float* fc2_b = (const float*)d_in[12];
  const float* w2    = (const float*)d_in[13];
  const float* b2    = (const float*)d_in[14];
  const float* g2    = (const float*)d_in[15];
  const float* bt2   = (const float*)d_in[16];

  char*  base = (char*)d_ws;
  size_t off  = 0;
  auto carve = [&](size_t bytes) -> void* {
    void* p = base + off;
    off = (off + bytes + 255) & ~(size_t)255;
    return p;
  };
  int*       fps_idx = (int*)carve((size_t)NGROUP * 4);
  _Float16*  w1h     = (_Float16*)carve((size_t)64 * 64 * 2);
  _Float16*  fc1h    = (_Float16*)carve((size_t)64 * 512 * 2);
  _Float16*  wqh     = (_Float16*)carve((size_t)512 * 512 * 2);
  _Float16*  wkh     = (_Float16*)carve((size_t)512 * 512 * 2);
  _Float16*  wvh     = (_Float16*)carve((size_t)512 * 512 * 2);
  _Float16*  fc2h    = (_Float16*)carve((size_t)512 * 64 * 2);
  _Float16*  w2h     = (_Float16*)carve((size_t)64 * 128 * 2);
  _Float16*  Afeat   = (_Float16*)carve((size_t)NROWS * CIN * 2);
  _Float16*  X1h     = (_Float16*)carve((size_t)NROWS * CIN * 2);
  _Float16*  Xbn     = (_Float16*)carve((size_t)NROWS * CIN * 2);
  _Float16*  Yh      = (_Float16*)carve((size_t)NROWS * COUT * 2);
  float*     stats   = (float*)carve((size_t)(128 + 256) * 4);
  float*     sb1     = (float*)carve((size_t)128 * 4);
  float*     sb2     = (float*)carve((size_t)256 * 4);
  float* stats1 = stats;
  float* stats2 = stats + 128;

  float* out_xyz  = (float*)d_out;                      // [8,1024,3]
  float* out_feat = (float*)d_out + (size_t)NGROUP * 3; // [8,1024,128]

  zero_f32<<<2, 256, 0, stream>>>(stats, 384);

  swizzle_weights<<<(64 * 64 + 255) / 256, 256, 0, stream>>>(w1, w1h, 64, 64);
  swizzle_weights<<<(512 * 64 + 255) / 256, 256, 0, stream>>>(fc1_w, fc1h, 512, 64);
  swizzle_weights<<<(512 * 512 + 255) / 256, 256, 0, stream>>>(wq, wqh, 512, 512);
  swizzle_weights<<<(512 * 512 + 255) / 256, 256, 0, stream>>>(wk, wkh, 512, 512);
  swizzle_weights<<<(512 * 512 + 255) / 256, 256, 0, stream>>>(wv, wvh, 512, 512);
  swizzle_weights<<<(64 * 512 + 255) / 256, 256, 0, stream>>>(fc2_w, fc2h, 64, 512);
  swizzle_weights<<<(128 * 64 + 255) / 256, 256, 0, stream>>>(w2, w2h, 128, 64);

  fps_kernel<<<BATCH, 256, 0, stream>>>(xyz, fps_idx, out_xyz);
  knn_gather_kernel<<<NGROUP, 256, 0, stream>>>(xyz, feat, fps_idx, Afeat);
  mlp1_kernel<<<NGROUP, 128, 0, stream>>>(Afeat, w1h, b1, X1h, stats1);
  bn_finalize<<<1, 64, 0, stream>>>(stats1, g1, bt1, sb1, sb1 + 64, 64,
                                    1.f / (float)NROWS);
  {
    size_t n = (size_t)NROWS * CIN;
    bn1_apply<<<(unsigned)((n + 255) / 256), 256, 0, stream>>>(X1h, sb1, sb1 + 64,
                                                               Xbn);
  }
  attn_kernel<<<NGROUP, 256, 0, stream>>>(Xbn, fc1h, fc1_b, wqh, wkh, wvh, fc2h,
                                          fc2_b, w2h, b2, Yh, stats2);
  bn_finalize<<<1, 128, 0, stream>>>(stats2, g2, bt2, sb2, sb2 + 128, 128,
                                     1.f / (float)NROWS);
  finalize_kernel<<<NGROUP, 128, 0, stream>>>(Yh, sb2, sb2 + 128, out_feat);
}
